// TransformerBlock_11793980195205
// MI455X (gfx1250) — compile-verified, matrix operand order checked by reading, not run on hardware
//
#include <hip/hip_runtime.h>
#include <math.h>

// ---------------- problem constants (match reference) ----------------
#define NNODES 50000
#define DCH    256
#define HEADS  4
#define CPH    64
#define NEDGES 800000
#define EPTOT  (NEDGES + NNODES)   // edges + self loops
#define NEG_SLOPE 0.2f
#define BN_EPS    1e-5f

typedef __attribute__((ext_vector_type(16))) _Float16 v16h;
typedef __attribute__((ext_vector_type(8)))  float    v8f;
typedef int i32x4_t __attribute__((vector_size(16)));   // matches builtin param type

union AFrag { v16h v; _Float16 h[16]; unsigned int u[8]; };
union H8    { _Float16 h[8]; uint4 q; };

// CDNA5 async memory->LDS path (gfx1250), guarded so either toolchain compiles
#ifndef USE_ASYNC_LDS
#if defined(__has_builtin)
#if __has_builtin(__builtin_amdgcn_global_load_async_to_lds_b128) && \
    __has_builtin(__builtin_amdgcn_s_wait_asynccnt)
#define USE_ASYNC_LDS 1
#endif
#endif
#endif
#ifndef USE_ASYNC_LDS
#define USE_ASYNC_LDS 0
#endif

// ---------------- helpers ----------------
__device__ __forceinline__ void atomicMaxF(float* addr, float v) {
    // ordered-int trick; segmax initialized to -inf
    if (v >= 0.f) atomicMax((int*)addr, __float_as_int(v));
    else          atomicMin((unsigned int*)addr, __float_as_uint(v));
}

// ---------------- generic fill ----------------
__global__ void fill_f32(float* __restrict__ p, float v, int n) {
    int i = blockIdx.x * blockDim.x + threadIdx.x;
    if (i < n) p[i] = v;
}

// ---------------- weight convert f32 -> f16 (4 matrices packed) ----------------
__global__ void convert_weights(const float* __restrict__ wl, const float* __restrict__ wr,
                                const float* __restrict__ w1, const float* __restrict__ w2,
                                _Float16* __restrict__ out) {
    int i = blockIdx.x * blockDim.x + threadIdx.x;
    const int per = DCH * DCH;
    if (i < per)            out[i] = (_Float16)wl[i];
    else if (i < 2 * per)   out[i] = (_Float16)wr[i - per];
    else if (i < 3 * per)   out[i] = (_Float16)w1[i - 2 * per];
    else if (i < 4 * per)   out[i] = (_Float16)w2[i - 3 * per];
}

// ---------------- WMMA GEMM: C[M,256] = A[M,256](f32->f16) @ B[256,256](f16) ----------------
// block = 128 threads (4 waves); block tile 64x64; each wave: 16 rows x four 16x16 WMMA tiles
__global__ __launch_bounds__(128) void gemm_wmma_f16(
    const float*    __restrict__ A,
    const _Float16* __restrict__ B,
    const float*    __restrict__ bias,   // per-column, may be null
    const float*    __restrict__ resid,  // [M,256] residual, may be null
    float*          __restrict__ Cout,
    int M, int applyRelu)
{
    __shared__ _Float16 As[64 * 32];   // 64 rows x 32 K-halves
    __shared__ _Float16 Bs[32 * 64];   // 32 K-rows x 64 cols

    const int m0   = blockIdx.x * 64;
    const int n0   = blockIdx.y * 64;
    const int tid  = threadIdx.x;
    const int wave = tid >> 5;
    const int lane = tid & 31;
    const int hl   = lane & 15;
    const int hi   = lane >> 4;

    v8f acc[4] = {};

#pragma unroll
    for (int k0 = 0; k0 < DCH; k0 += 32) {
        // stage B: 256 slots of 8 halves (16B each) -> ASYNCcnt-tracked LDS DMA
#if USE_ASYNC_LDS
        for (int s = tid; s < 256; s += 128) {
            int row = s >> 3;            // 0..31 (K)
            int col = (s & 7) * 8;       // 0..56
            __builtin_amdgcn_global_load_async_to_lds_b128(
                (__attribute__((address_space(1))) i32x4_t*)(void*)
                    (B + (size_t)(k0 + row) * DCH + n0 + col),
                (__attribute__((address_space(3))) i32x4_t*)(void*)
                    (&Bs[row * 64 + col]),
                0, 0);
        }
#else
        for (int s = tid; s < 256; s += 128) {
            int row = s >> 3;
            int col = (s & 7) * 8;
            *(uint4*)&Bs[row * 64 + col] =
                *(const uint4*)(B + (size_t)(k0 + row) * DCH + n0 + col);
        }
#endif
        // stage A (convert f32->f16): 256 slots of 8 halves
        for (int s = tid; s < 256; s += 128) {
            int row = s >> 2;            // 0..63
            int col = (s & 3) * 8;       // 0,8,16,24
            int gr  = m0 + row;
            H8 t;
            if (gr < M) {
                const float4* ap = (const float4*)(A + (size_t)gr * DCH + k0 + col);
                float4 f0 = ap[0], f1 = ap[1];
                t.h[0] = (_Float16)f0.x; t.h[1] = (_Float16)f0.y;
                t.h[2] = (_Float16)f0.z; t.h[3] = (_Float16)f0.w;
                t.h[4] = (_Float16)f1.x; t.h[5] = (_Float16)f1.y;
                t.h[6] = (_Float16)f1.z; t.h[7] = (_Float16)f1.w;
            } else {
                t.q = make_uint4(0u, 0u, 0u, 0u);
            }
            *(uint4*)&As[row * 32 + col] = t.q;
        }
#if USE_ASYNC_LDS
        __builtin_amdgcn_s_wait_asynccnt(0);   // own async LDS writes done
#endif
        __syncthreads();                        // all waves' writes visible

        // A fragment: ISA 7.12.2 16-bit A 16x32 layout (row per half-lane, K split by hi)
        AFrag a;
#pragma unroll
        for (int i = 0; i < 8; ++i) {
            int kb = (i < 4 ? 2 * i : 16 + 2 * (i - 4)) + hi * 8;
            a.u[i] = *(const unsigned int*)&As[(wave * 16 + hl) * 32 + kb];
        }
#pragma unroll
        for (int nt = 0; nt < 4; ++nt) {
            AFrag b;   // B fragment: mirrored layout, N striped across half-lanes
#pragma unroll
            for (int i = 0; i < 8; ++i) {
                int kb = (i < 4 ? 2 * i : 16 + 2 * (i - 4)) + hi * 8;
                b.h[2 * i]     = Bs[kb * 64 + nt * 16 + hl];
                b.h[2 * i + 1] = Bs[(kb + 1) * 64 + nt * 16 + hl];
            }
            acc[nt] = __builtin_amdgcn_wmma_f32_16x16x32_f16(
                false, a.v, false, b.v, (short)0, acc[nt], false, false);
        }
        __syncthreads();
    }

    // epilogue: C layout — lane(0-15): N=lane, M=j; lane(16-31): N=lane-16, M=8+j
#pragma unroll
    for (int nt = 0; nt < 4; ++nt) {
        int col = n0 + nt * 16 + hl;
        float bv = bias ? bias[col] : 0.f;
#pragma unroll
        for (int j = 0; j < 8; ++j) {
            int row = m0 + wave * 16 + hi * 8 + j;
            if (row < M) {
                float v = acc[nt][j] + bv;
                if (resid)     v += resid[(size_t)row * DCH + col];
                if (applyRelu) v  = v > 0.f ? v : 0.f;
                Cout[(size_t)row * DCH + col] = v;
            }
        }
    }
}

// ---------------- pass A: edge logits + segment max (wave per edge) ----------------
__global__ __launch_bounds__(256) void edge_logits_max(
    const int* __restrict__ ei, const float* __restrict__ xl, const float* __restrict__ xr,
    const float* __restrict__ att, float* __restrict__ logits, float* __restrict__ segmax)
{
    const int lane  = threadIdx.x & 31;
    const int wib   = threadIdx.x >> 5;
    const long long stride = (long long)gridDim.x * 8;
    const int ch0   = lane * 8;
    const int head  = lane >> 3;

    float av[8];
#pragma unroll
    for (int j = 0; j < 8; ++j) av[j] = att[head * CPH + (lane & 7) * 8 + j];

    for (long long e = (long long)blockIdx.x * 8 + wib; e < EPTOT; e += stride) {
        int src, dst;
        if (e < NEDGES) { src = ei[e]; dst = ei[NEDGES + e]; }
        else            { src = dst = (int)(e - NEDGES); }

        long long en = e + stride;           // prefetch next iteration's rows
        if (en < EPTOT) {
            int sn, dn;
            if (en < NEDGES) { sn = ei[en]; dn = ei[NEDGES + en]; }
            else             { sn = dn = (int)(en - NEDGES); }
            __builtin_prefetch(&xl[(size_t)sn * DCH + ch0], 0, 1);
            __builtin_prefetch(&xr[(size_t)dn * DCH + ch0], 0, 1);
        }

        const float4* xa = (const float4*)(xl + (size_t)src * DCH + ch0);
        const float4* xb = (const float4*)(xr + (size_t)dst * DCH + ch0);
        float4 a0 = xa[0], a1 = xa[1], b0 = xb[0], b1 = xb[1];
        float f[8] = { a0.x + b0.x, a0.y + b0.y, a0.z + b0.z, a0.w + b0.w,
                       a1.x + b1.x, a1.y + b1.y, a1.z + b1.z, a1.w + b1.w };
        float p = 0.f;
#pragma unroll
        for (int j = 0; j < 8; ++j) {
            float t = f[j];
            t = t >= 0.f ? t : NEG_SLOPE * t;
            p += t * av[j];
        }
        p += __shfl_xor(p, 1, 8);
        p += __shfl_xor(p, 2, 8);
        p += __shfl_xor(p, 4, 8);
        if ((lane & 7) == 0) {
            logits[e * HEADS + head] = p;
            atomicMaxF(&segmax[(size_t)dst * HEADS + head], p);
        }
    }
}

// ---------------- pass B: ex = exp(logit - max), segment sum ----------------
__global__ void edge_softmax_sum(const int* __restrict__ ei, float* __restrict__ logits_ex,
                                 const float* __restrict__ segmax, float* __restrict__ segsum)
{
    long long e = (long long)blockIdx.x * blockDim.x + threadIdx.x;
    if (e >= EPTOT) return;
    int dst = (e < NEDGES) ? ei[NEDGES + e] : (int)(e - NEDGES);
    float4 lg = *(float4*)(logits_ex + e * HEADS);
    const float* sm = segmax + (size_t)dst * HEADS;
    float4 ex;
    ex.x = __expf(lg.x - sm[0]);
    ex.y = __expf(lg.y - sm[1]);
    ex.z = __expf(lg.z - sm[2]);
    ex.w = __expf(lg.w - sm[3]);
    *(float4*)(logits_ex + e * HEADS) = ex;
    atomicAdd(&segsum[(size_t)dst * HEADS + 0], ex.x);
    atomicAdd(&segsum[(size_t)dst * HEADS + 1], ex.y);
    atomicAdd(&segsum[(size_t)dst * HEADS + 2], ex.z);
    atomicAdd(&segsum[(size_t)dst * HEADS + 3], ex.w);
}

// ---------------- pass C: weighted scatter-sum of source messages ----------------
__global__ __launch_bounds__(256) void edge_scatter(
    const int* __restrict__ ei, const float* __restrict__ xl, const float* __restrict__ ex,
    const float* __restrict__ segsum, float* __restrict__ attn)
{
    const int lane = threadIdx.x & 31;
    const int wib  = threadIdx.x >> 5;
    const long long stride = (long long)gridDim.x * 8;
    const int ch0  = lane * 8;
    const int head = lane >> 3;

    for (long long e = (long long)blockIdx.x * 8 + wib; e < EPTOT; e += stride) {
        int src, dst;
        if (e < NEDGES) { src = ei[e]; dst = ei[NEDGES + e]; }
        else            { src = dst = (int)(e - NEDGES); }

        long long en = e + stride;
        if (en < EPTOT) {
            int sn = (en < NEDGES) ? ei[en] : (int)(en - NEDGES);
            __builtin_prefetch(&xl[(size_t)sn * DCH + ch0], 0, 1);
        }

        float alpha = ex[e * HEADS + head] /
                      (segsum[(size_t)dst * HEADS + head] + 1e-16f);
        const float4* xs = (const float4*)(xl + (size_t)src * DCH + ch0);
        float4 m0 = xs[0], m1 = xs[1];
        float* ap = attn + (size_t)dst * DCH + ch0;
        atomicAdd(ap + 0, alpha * m0.x);
        atomicAdd(ap + 1, alpha * m0.y);
        atomicAdd(ap + 2, alpha * m0.z);
        atomicAdd(ap + 3, alpha * m0.w);
        atomicAdd(ap + 4, alpha * m1.x);
        atomicAdd(ap + 5, alpha * m1.y);
        atomicAdd(ap + 6, alpha * m1.z);
        atomicAdd(ap + 7, alpha * m1.w);
    }
}

// ---------------- residual 1: r1 = attn + bias + x ----------------
__global__ void residual1(const float* __restrict__ attn, const float* __restrict__ bias,
                          const float* __restrict__ x, float* __restrict__ out)
{
    size_t i = (size_t)blockIdx.x * blockDim.x + threadIdx.x;
    if (i >= (size_t)NNODES * DCH) return;
    int c = (int)(i & (DCH - 1));
    out[i] = attn[i] + bias[c] + x[i];
}

// ---------------- batchnorm (training-mode batch stats over axis 0) ----------------
__global__ __launch_bounds__(256) void bn_stats(const float* __restrict__ src,
                                                float* __restrict__ sum, float* __restrict__ sumsq)
{
    int c = threadIdx.x;            // column 0..255
    float s = 0.f, s2 = 0.f;
    for (int r = blockIdx.x; r < NNODES; r += gridDim.x) {
        float v = src[(size_t)r * DCH + c];
        s += v; s2 += v * v;
    }
    atomicAdd(&sum[c], s);
    atomicAdd(&sumsq[c], s2);
}

__global__ void bn_finalize(const float* __restrict__ sum, const float* __restrict__ sumsq,
                            float* __restrict__ mean, float* __restrict__ invstd)
{
    int c = threadIdx.x;
    float m   = sum[c] / (float)NNODES;
    float var = sumsq[c] / (float)NNODES - m * m;   // biased, as reference
    mean[c]   = m;
    invstd[c] = rsqrtf(var + BN_EPS);
}

__global__ void bn_apply(const float* __restrict__ src, const float* __restrict__ mean,
                         const float* __restrict__ invstd, const float* __restrict__ gamma,
                         const float* __restrict__ beta, float* __restrict__ dst)
{
    size_t i = (size_t)blockIdx.x * blockDim.x + threadIdx.x;
    if (i >= (size_t)NNODES * DCH) return;
    int c = (int)(i & (DCH - 1));
    dst[i] = gamma[c] * (src[i] - mean[c]) * invstd[c] + beta[c];
}

// ---------------- launch ----------------
extern "C" void kernel_launch(void* const* d_in, const int* in_sizes, int n_in,
                              void* d_out, int out_size, void* d_ws, size_t ws_size,
                              hipStream_t stream)
{
    const float* x     = (const float*)d_in[0];
    const int*   ei    = (const int*)d_in[1];
    const float* W_l   = (const float*)d_in[2];
    const float* W_r   = (const float*)d_in[3];
    const float* att   = (const float*)d_in[4];
    const float* bias  = (const float*)d_in[5];
    const float* W1    = (const float*)d_in[6];
    const float* b1    = (const float*)d_in[7];
    const float* W2    = (const float*)d_in[8];
    const float* b2    = (const float*)d_in[9];
    const float* g1    = (const float*)d_in[10];
    const float* be1   = (const float*)d_in[11];
    const float* g2    = (const float*)d_in[12];
    const float* be2   = (const float*)d_in[13];

    char* ws = (char*)d_ws;
    size_t off = 0;
    auto take = [&](size_t bytes) -> void* {
        void* p = ws + off;
        off += (bytes + 255) & ~(size_t)255;
        return p;
    };
    const size_t ND = (size_t)NNODES * DCH;
    _Float16* wh     = (_Float16*)take(4 * (size_t)DCH * DCH * sizeof(_Float16));
    float*    xl     = (float*)take(ND * sizeof(float));                 // later reused as FFN1 out
    float*    xr     = (float*)take(ND * sizeof(float));                 // later attn accum, then r2
    float*    logits = (float*)take((size_t)EPTOT * HEADS * sizeof(float)); // logits then ex (in place)
    float*    segmax = (float*)take((size_t)NNODES * HEADS * sizeof(float));
    float*    segsum = (float*)take((size_t)NNODES * HEADS * sizeof(float));
    float*    hbuf   = (float*)take(ND * sizeof(float));                 // r1 then h (in place)
    float*    stats  = (float*)take(8 * DCH * sizeof(float));            // sum,sq,mean,istd x2 sets
    float*    tbuf   = xl;   // FFN1 output (xl dead after pass C)
    float*    r2     = xr;   // FFN2+resid output (attn dead after residual1)

    const int DD = DCH * DCH;
    dim3 gemmGrid((NNODES + 63) / 64, DCH / 64);

    convert_weights<<<(4 * DD + 255) / 256, 256, 0, stream>>>(W_l, W_r, W1, W2, wh);

    // xl = x @ W_l ; xr = x @ W_r   (WMMA f16 -> f32 accum)
    gemm_wmma_f16<<<gemmGrid, 128, 0, stream>>>(x, wh,        nullptr, nullptr, xl, NNODES, 0);
    gemm_wmma_f16<<<gemmGrid, 128, 0, stream>>>(x, wh + DD,   nullptr, nullptr, xr, NNODES, 0);

    // segment softmax scratch init
    fill_f32<<<(NNODES * HEADS + 255) / 256, 256, 0, stream>>>(segmax, -INFINITY, NNODES * HEADS);
    fill_f32<<<(NNODES * HEADS + 255) / 256, 256, 0, stream>>>(segsum, 0.f, NNODES * HEADS);

    edge_logits_max<<<4096, 256, 0, stream>>>(ei, xl, xr, att, logits, segmax);

    // attn accumulator (reuses xr) must be zeroed after pass A consumed xr
    fill_f32<<<(int)((ND + 255) / 256), 256, 0, stream>>>(xr, 0.f, (int)ND);

    edge_softmax_sum<<<(EPTOT + 255) / 256, 256, 0, stream>>>(ei, logits, segmax, segsum);
    edge_scatter<<<4096, 256, 0, stream>>>(ei, xl, logits, segsum, xr);

    // r1 = attn + bias + x ; BN1
    residual1<<<(int)((ND + 255) / 256), 256, 0, stream>>>(xr, bias, x, hbuf);
    fill_f32<<<2, 256, 0, stream>>>(stats, 0.f, 2 * DCH);
    bn_stats<<<256, 256, 0, stream>>>(hbuf, stats, stats + DCH);
    bn_finalize<<<1, 256, 0, stream>>>(stats, stats + DCH, stats + 2 * DCH, stats + 3 * DCH);
    bn_apply<<<(int)((ND + 255) / 256), 256, 0, stream>>>(hbuf, stats + 2 * DCH, stats + 3 * DCH,
                                                          g1, be1, hbuf);

    // FFN: t = relu(h @ W1 + b1) ; r2 = t @ W2 + b2 + h
    gemm_wmma_f16<<<gemmGrid, 128, 0, stream>>>(hbuf, wh + 2 * DD, b1, nullptr, tbuf, NNODES, 1);
    gemm_wmma_f16<<<gemmGrid, 128, 0, stream>>>(tbuf, wh + 3 * DD, b2, hbuf,    r2,   NNODES, 0);

    // BN2 -> d_out
    fill_f32<<<2, 256, 0, stream>>>(stats + 4 * DCH, 0.f, 2 * DCH);
    bn_stats<<<256, 256, 0, stream>>>(r2, stats + 4 * DCH, stats + 5 * DCH);
    bn_finalize<<<1, 256, 0, stream>>>(stats + 4 * DCH, stats + 5 * DCH,
                                       stats + 6 * DCH, stats + 7 * DCH);
    bn_apply<<<(int)((ND + 255) / 256), 256, 0, stream>>>(r2, stats + 6 * DCH, stats + 7 * DCH,
                                                          g2, be2, (float*)d_out);
}
